// AdvancedVectorMemory_55722905699063
// MI455X (gfx1250) — matrix operand order repaired and checked by reading, not compiled, not run on hardware
//
#include <hip/hip_runtime.h>
#include <hip/hip_bf16.h>

typedef __attribute__((ext_vector_type(16))) _Float16 v16h;
typedef __attribute__((ext_vector_type(8)))  _Float16 v8h;
typedef __attribute__((ext_vector_type(8)))  float    v8f;
typedef int v4i __attribute__((ext_vector_type(4)));

union AFrag { v16h v; v8h h[2]; };

#define WMMA_F16(a,b,c) \
  __builtin_amdgcn_wmma_f32_16x16x32_f16(false,(a),false,(b),(short)0,(c),false,false)

#define B_SZ   2
#define S_LEN  2048
#define M_LEN  4096
#define DMODEL 1024
#define DMEM   768
#define HEADS  16
#define DH     64

// Async global->LDS staging (gfx1250): probe-guarded so the file compiles on
// toolchains that do not declare these builtins (falls back to direct copies).
#if defined(__HIP_DEVICE_COMPILE__) && \
    __has_builtin(__builtin_amdgcn_global_load_async_to_lds_b128) && \
    __has_builtin(__builtin_amdgcn_s_wait_asynccnt)
#define HAVE_ASYNC_LDS 1
// builtin signature: (int4 addrspace(1)* src, int4 addrspace(3)* dst, imm off, imm cpol)
typedef v4i __attribute__((address_space(1)))* gas_v4i;
typedef v4i __attribute__((address_space(3)))* las_v4i;
#else
#define HAVE_ASYNC_LDS 0
#endif

// ---------------------------------------------------------------------------
// f32 -> f16 with optional output leading-dimension (for writing into concat)
// ---------------------------------------------------------------------------
__global__ void k_cvt2d(const float* __restrict__ in, _Float16* __restrict__ out,
                        long long n, int cols, int oldd) {
  long long i = (long long)blockIdx.x * blockDim.x + threadIdx.x;
  long long stride = (long long)gridDim.x * blockDim.x;
  for (; i < n; i += stride) {
    long long r = i / cols;
    int c = (int)(i - r * cols);
    out[r * (long long)oldd + c] = (_Float16)in[i];
  }
}

// ---------------------------------------------------------------------------
// WMMA GEMM: C[m,n] = act( sum_k A[m,k]*W[n,k] + bias[n] )
// A f16 [M,K] lda, W f16 [N,K] (torch Linear layout), C f16 ldc.
// Register-blocked: one wave computes a 32x64 macro-tile:
//   2 A-frags x 4 B-frags -> 8 v_wmma per 32-wide k-step (A and B both reused).
// act: 0=none, 1=SiLU.
// ---------------------------------------------------------------------------
__global__ void __launch_bounds__(256)
k_gemm_f16(const _Float16* __restrict__ A, int lda,
           const _Float16* __restrict__ W, int K,
           const float* __restrict__ bias,
           _Float16* __restrict__ C, int ldc,
           int tilesN4, int act) {
  const int wave = threadIdx.x >> 5;
  const int lane = threadIdx.x & 31;
  const int hf   = lane >> 4;     // half-wave select
  const int l16  = lane & 15;
  const int tile = blockIdx.x * 8 + wave;
  const int tm = tile / tilesN4;        // 32-row group
  const int tn = tile - tm * tilesN4;   // 64-col group
  const int m0 = tm * 32, n0 = tn * 64;

  const _Float16* A0 = A + (long long)(m0 + l16) * lda;
  const _Float16* A1 = A0 + (long long)16 * lda;
  const _Float16* W0 = W + (long long)(n0 + l16) * K;

  v8f acc[2][4];
  #pragma unroll
  for (int j = 0; j < 4; ++j) {
    const float bval = bias[n0 + j * 16 + l16];
    #pragma unroll
    for (int r = 0; r < 8; ++r) { acc[0][j][r] = bval; acc[1][j][r] = bval; }
  }

  for (int k0 = 0; k0 < K; k0 += 32) {
    // A frag (16x32): lane half holds K = [k0+hf*16,+16) contiguous
    AFrag a0, a1;
    a0.h[0] = *(const v8h*)(A0 + k0 + hf * 16);
    a0.h[1] = *(const v8h*)(A0 + k0 + hf * 16 + 8);
    a1.h[0] = *(const v8h*)(A1 + k0 + hf * 16);
    a1.h[1] = *(const v8h*)(A1 + k0 + hf * 16 + 8);
    #pragma unroll
    for (int j = 0; j < 4; ++j) {
      // B frag (32x16): lane = N, 16 contiguous K values per lane-half
      v16h bfr = *(const v16h*)(W0 + (long long)j * 16 * K + k0 + hf * 16);
      acc[0][j] = WMMA_F16(a0.v, bfr, acc[0][j]);
      acc[1][j] = WMMA_F16(a1.v, bfr, acc[1][j]);
    }
  }

  // C layout: VGPR r -> row + r + hf*8, col = base + l16
  #pragma unroll
  for (int i = 0; i < 2; ++i) {
    _Float16* Crow = C + (long long)(m0 + i * 16 + hf * 8) * ldc + n0 + l16;
    #pragma unroll
    for (int j = 0; j < 4; ++j) {
      #pragma unroll
      for (int r = 0; r < 8; ++r) {
        float x = acc[i][j][r];
        if (act == 1) x = x / (1.f + __expf(-x));   // SiLU
        Crow[(long long)r * ldc + j * 16] = (_Float16)x;
      }
    }
  }
}

// ---------------------------------------------------------------------------
// Flash attention: block = 4 waves, each wave one 16-row s-tile of one (b,h).
// K/V chunks of 32 memory rows staged in LDS, via async global->LDS copies
// (ASYNCcnt path) when available, else direct copies.
// ---------------------------------------------------------------------------
__global__ void __launch_bounds__(128)
k_attn(const _Float16* __restrict__ Q, const _Float16* __restrict__ Kp,
       const _Float16* __restrict__ Vp, _Float16* __restrict__ R) {
  __shared__ __attribute__((aligned(64))) _Float16 Ks [32][64];     // [mem][d]
  __shared__ __attribute__((aligned(64))) _Float16 Vst[32][64];     // staging [mem][d]
  __shared__ __attribute__((aligned(64))) _Float16 Vs [64][32];     // transposed [d][mem]
  __shared__ __attribute__((aligned(64))) _Float16 Ps [4][16][32];  // per-wave P

  const int tid  = threadIdx.x;
  const int wave = tid >> 5;
  const int lane = tid & 31;
  const int hf   = lane >> 4;
  const int l16  = lane & 15;

  const int id  = blockIdx.x;
  const int st4 = id & 31;          // S/64 groups
  const int h   = (id >> 5) & 15;
  const int b   = id >> 9;
  const int s0  = st4 * 64 + wave * 16;

  const _Float16* qrow = Q + ((long long)(b * S_LEN + s0 + l16) * DMODEL + h * DH);
  AFrag aq0, aq1;
  aq0.h[0] = *(const v8h*)(qrow + hf * 16);
  aq0.h[1] = *(const v8h*)(qrow + hf * 16 + 8);
  aq1.h[0] = *(const v8h*)(qrow + 32 + hf * 16);
  aq1.h[1] = *(const v8h*)(qrow + 40 + hf * 16);

  v8f o0 = {}, o1 = {}, o2 = {}, o3 = {};
  float mrow[8], lrow[8];
  #pragma unroll
  for (int r = 0; r < 8; ++r) { mrow[r] = -3.0e30f; lrow[r] = 0.f; }

  const int ldrow = tid >> 2;   // 0..31: memory row within chunk
  const int ldseg = tid & 3;    // 16-half segment of d

  for (int m0 = 0; m0 < M_LEN; m0 += 32) {
    __syncthreads();   // previous chunk fully consumed before restaging

#if HAVE_ASYNC_LDS
    {
      // 4KB per array; each lane moves 16B per op, 128 lanes -> 2KB/op, 2 ops.
      #pragma unroll
      for (int s = 0; s < 2; ++s) {
        const int o    = tid * 16 + s * 2048;   // byte offset within chunk
        const int mrw  = o >> 7;                // 128B per memory row
        const int db   = o & 127;
        const char* gK = (const char*)(Kp +
            ((long long)(b * M_LEN + m0 + mrw) * DMODEL + h * DH)) + db;
        const char* gV = (const char*)(Vp +
            ((long long)(b * M_LEN + m0 + mrw) * DMODEL + h * DH)) + db;
        __builtin_amdgcn_global_load_async_to_lds_b128(
            (gas_v4i)(void*)gK,
            (las_v4i)(void*)((char*)&Ks[0][0] + o), 0, 0);
        __builtin_amdgcn_global_load_async_to_lds_b128(
            (gas_v4i)(void*)gV,
            (las_v4i)(void*)((char*)&Vst[0][0] + o), 0, 0);
      }
      __builtin_amdgcn_s_wait_asynccnt(0);
    }
    __syncthreads();
    {
      // LDS->LDS transpose of the V chunk so the PV B-operand is contiguous
      v16h tmp = *(const v16h*)&Vst[ldrow][ldseg * 16];
      #pragma unroll
      for (int i = 0; i < 16; ++i) Vs[ldseg * 16 + i][ldrow] = tmp[i];
    }
#else
    {
      const _Float16* src =
          Kp + ((long long)(b * M_LEN + m0 + ldrow) * DMODEL + h * DH + ldseg * 16);
      *(v16h*)&Ks[ldrow][ldseg * 16] = *(const v16h*)src;
    }
    {
      const _Float16* src =
          Vp + ((long long)(b * M_LEN + m0 + ldrow) * DMODEL + h * DH + ldseg * 16);
      v16h tmp = *(const v16h*)src;
      #pragma unroll
      for (int i = 0; i < 16; ++i) Vs[ldseg * 16 + i][ldrow] = tmp[i];
    }
#endif
    __syncthreads();

    // scores: two 16-wide memory tiles, K-dim = d split 0..31 / 32..63
    v8f s_t0 = {}, s_t1 = {};
    {
      v16h bk0 = *(const v16h*)&Ks[l16][hf * 16];
      v16h bk1 = *(const v16h*)&Ks[l16][32 + hf * 16];
      s_t0 = WMMA_F16(aq0.v, bk0, s_t0);
      s_t0 = WMMA_F16(aq1.v, bk1, s_t0);
      v16h bk2 = *(const v16h*)&Ks[16 + l16][hf * 16];
      v16h bk3 = *(const v16h*)&Ks[16 + l16][32 + hf * 16];
      s_t1 = WMMA_F16(aq0.v, bk2, s_t1);
      s_t1 = WMMA_F16(aq1.v, bk3, s_t1);
    }

    const float scale = 0.125f;   // 1/sqrt(64)
    #pragma unroll
    for (int r = 0; r < 8; ++r) { s_t0[r] *= scale; s_t1[r] *= scale; }

    // online softmax; (r, lane-half) owns row r + hf*8, cols across 16 lanes
    #pragma unroll
    for (int r = 0; r < 8; ++r) {
      float cm = fmaxf(s_t0[r], s_t1[r]);
      #pragma unroll
      for (int m = 8; m >= 1; m >>= 1) cm = fmaxf(cm, __shfl_xor(cm, m, 16));
      float newm  = fmaxf(mrow[r], cm);
      float alpha = __expf(mrow[r] - newm);
      float p0 = __expf(s_t0[r] - newm);
      float p1 = __expf(s_t1[r] - newm);
      float rs = p0 + p1;
      #pragma unroll
      for (int m = 8; m >= 1; m >>= 1) rs += __shfl_xor(rs, m, 16);
      lrow[r] = lrow[r] * alpha + rs;
      mrow[r] = newm;
      o0[r] *= alpha; o1[r] *= alpha; o2[r] *= alpha; o3[r] *= alpha;
      const int rw = r + hf * 8;
      Ps[wave][rw][l16]      = (_Float16)p0;
      Ps[wave][rw][16 + l16] = (_Float16)p1;
    }
    __builtin_amdgcn_wave_barrier();  // keep same-wave LDS store->load order

    // P re-layout: C-frag -> A-frag via LDS
    AFrag ap;
    ap.h[0] = *(const v8h*)&Ps[wave][l16][hf * 16];
    ap.h[1] = *(const v8h*)&Ps[wave][l16][hf * 16 + 8];

    // PV: B frag lane = d column, 16 contiguous mem values per lane-half
    v16h bv0 = *(const v16h*)&Vs[ 0 + l16][hf * 16];
    v16h bv1 = *(const v16h*)&Vs[16 + l16][hf * 16];
    v16h bv2 = *(const v16h*)&Vs[32 + l16][hf * 16];
    v16h bv3 = *(const v16h*)&Vs[48 + l16][hf * 16];
    o0 = WMMA_F16(ap.v, bv0, o0);
    o1 = WMMA_F16(ap.v, bv1, o1);
    o2 = WMMA_F16(ap.v, bv2, o2);
    o3 = WMMA_F16(ap.v, bv3, o3);
  }

  #pragma unroll
  for (int r = 0; r < 8; ++r) {
    float inv = 1.f / lrow[r];
    long long srow = (long long)(b * S_LEN + s0 + r + hf * 8) * DMODEL + h * DH + l16;
    R[srow +  0] = (_Float16)(o0[r] * inv);
    R[srow + 16] = (_Float16)(o1[r] * inv);
    R[srow + 32] = (_Float16)(o2[r] * inv);
    R[srow + 48] = (_Float16)(o3[r] * inv);
  }
}

// ---------------------------------------------------------------------------
__global__ void __launch_bounds__(256)
k_gate(const _Float16* __restrict__ G1, const float* __restrict__ Wg2,
       const float* __restrict__ bg2, float* __restrict__ gate) {
  const int row  = blockIdx.x * 8 + (threadIdx.x >> 5);
  const int lane = threadIdx.x & 31;
  const _Float16* g = G1 + (long long)row * DMODEL;
  float s = 0.f;
  for (int j = lane; j < DMODEL; j += 32) s += (float)g[j] * Wg2[j];
  #pragma unroll
  for (int m = 16; m >= 1; m >>= 1) s += __shfl_xor(s, m, 32);
  if (lane == 0) gate[row] = 1.f / (1.f + __expf(-(s + bg2[0])));
}

// ---------------------------------------------------------------------------
__global__ void k_final(const float* __restrict__ query,
                        const _Float16* __restrict__ concatb,
                        const float* __restrict__ gate,
                        float* __restrict__ out, long long n) {
  long long i = (long long)blockIdx.x * blockDim.x + threadIdx.x;
  if (i >= n) return;
  long long row = i >> 10;
  int c = (int)(i & 1023);
  out[i] = query[i] + gate[row] * (float)concatb[row * 2048 + 1024 + c];
}

// ---------------------------------------------------------------------------
extern "C" void kernel_launch(void* const* d_in, const int* in_sizes, int n_in,
                              void* d_out, int out_size, void* d_ws, size_t ws_size,
                              hipStream_t stream) {
  const float* query = (const float*)d_in[0];
  const float* mkeys = (const float*)d_in[1];
  const float* mvals = (const float*)d_in[2];
  const float* Wq = (const float*)d_in[3];  const float* bq = (const float*)d_in[4];
  const float* Wk = (const float*)d_in[5];  const float* bk = (const float*)d_in[6];
  const float* Wv = (const float*)d_in[7];  const float* bv = (const float*)d_in[8];
  const float* Wo = (const float*)d_in[9];  const float* bo = (const float*)d_in[10];
  const float* Wg1 = (const float*)d_in[11]; const float* bg1 = (const float*)d_in[12];
  const float* Wg2 = (const float*)d_in[13]; const float* bg2 = (const float*)d_in[14];
  float* out = (float*)d_out;

  const long long NQ = (long long)B_SZ * S_LEN;   // 4096 query rows
  const long long NM = (long long)B_SZ * M_LEN;   // 8192 memory rows

  char* p = (char*)d_ws;
  auto carve = [&](size_t bytes) {
    void* q = (void*)p;
    p += (bytes + 255) & ~(size_t)255;
    return q;
  };
  _Float16* concatb = (_Float16*)carve((size_t)NQ * 2048 * 2);  // [query | retrieved]
  _Float16* Xk   = (_Float16*)carve((size_t)NM * DMEM * 2);
  _Float16* Xv   = (_Float16*)carve((size_t)NM * DMEM * 2);
  _Float16* Wq16 = (_Float16*)carve((size_t)DMODEL * DMODEL * 2);
  _Float16* Wk16 = (_Float16*)carve((size_t)DMODEL * DMEM * 2);
  _Float16* Wv16 = (_Float16*)carve((size_t)DMODEL * DMEM * 2);
  _Float16* Wo16 = (_Float16*)carve((size_t)DMODEL * DMODEL * 2);
  _Float16* Wg16 = (_Float16*)carve((size_t)DMODEL * 2048 * 2);
  _Float16* Q16  = (_Float16*)carve((size_t)NQ * DMODEL * 2);
  _Float16* K16  = (_Float16*)carve((size_t)NM * DMODEL * 2);
  _Float16* V16  = (_Float16*)carve((size_t)NM * DMODEL * 2);
  _Float16* R16  = (_Float16*)carve((size_t)NQ * DMODEL * 2);
  _Float16* G1   = (_Float16*)carve((size_t)NQ * DMODEL * 2);
  float*    gate = (float*)carve((size_t)NQ * 4);

  auto cvt = [&](const float* in, _Float16* o, long long n, int cols, int oldd) {
    int blocks = (int)((n + 255) / 256);
    k_cvt2d<<<blocks, 256, 0, stream>>>(in, o, n, cols, oldd);
  };

  cvt(query, concatb, NQ * DMODEL, DMODEL, 2048);
  cvt(mkeys, Xk, NM * DMEM, DMEM, DMEM);
  cvt(mvals, Xv, NM * DMEM, DMEM, DMEM);
  cvt(Wq, Wq16, (long long)DMODEL * DMODEL, DMODEL, DMODEL);
  cvt(Wk, Wk16, (long long)DMODEL * DMEM, DMEM, DMEM);
  cvt(Wv, Wv16, (long long)DMODEL * DMEM, DMEM, DMEM);
  cvt(Wo, Wo16, (long long)DMODEL * DMODEL, DMODEL, DMODEL);
  cvt(Wg1, Wg16, (long long)DMODEL * 2048, 2048, 2048);

  const int tilesN4 = DMODEL / 64;   // 16 64-col groups

  // Q = concat[:, :1024] @ Wq^T + bq    (128*16 macro-tiles / 8 waves -> 256 blocks)
  k_gemm_f16<<<(int)(NQ / 32 * tilesN4 / 8), 256, 0, stream>>>(
      concatb, 2048, Wq16, DMODEL, bq, Q16, DMODEL, tilesN4, 0);
  // K = mkeys @ Wk^T + bk               (256*16 / 8 -> 512 blocks)
  k_gemm_f16<<<(int)(NM / 32 * tilesN4 / 8), 256, 0, stream>>>(
      Xk, DMEM, Wk16, DMEM, bk, K16, DMODEL, tilesN4, 0);
  // V = mvals @ Wv^T + bv
  k_gemm_f16<<<(int)(NM / 32 * tilesN4 / 8), 256, 0, stream>>>(
      Xv, DMEM, Wv16, DMEM, bv, V16, DMODEL, tilesN4, 0);

  // flash attention -> R16
  k_attn<<<B_SZ * HEADS * (S_LEN / 64), 128, 0, stream>>>(Q16, K16, V16, R16);

  // output projection written into concat[:, 1024:]
  k_gemm_f16<<<(int)(NQ / 32 * tilesN4 / 8), 256, 0, stream>>>(
      R16, DMODEL, Wo16, DMODEL, bo, concatb + 1024, 2048, tilesN4, 0);

  // gate MLP layer 1 with SiLU
  k_gemm_f16<<<(int)(NQ / 32 * tilesN4 / 8), 256, 0, stream>>>(
      concatb, 2048, Wg16, 2048, bg1, G1, DMODEL, tilesN4, 1);

  // gate scalar + sigmoid
  k_gate<<<(int)(NQ / 8), 256, 0, stream>>>(G1, Wg2, bg2, gate);

  // out = query + gate * retrieved
  const long long n = NQ * DMODEL;
  k_final<<<(int)((n + 255) / 256), 256, 0, stream>>>(query, concatb, gate, out, n);
}
